// Block_89773406421284
// MI455X (gfx1250) — matrix operand order repaired and checked by reading, not compiled, not run on hardware
//
#include <hip/hip_runtime.h>
#include <hip/hip_fp16.h>

// RWKV-v4 block for MI455X (gfx1250, wave32).
// All 7 GEMMs run on v_wmma_f32_16x16x32_f16 (f16 in / f32 accumulate).
// WKV scan stays f32 scalar (sequential over T, latency bound).
// GEMM wave tile: 32(M) x 64(N), manually double-buffered K pipeline.

typedef __attribute__((ext_vector_type(16))) _Float16 v16h;
typedef __attribute__((ext_vector_type(8)))  _Float16 half8;
typedef __attribute__((ext_vector_type(8)))  float    v8f;

#define PP 2
#define BB 8
#define TT 1024
#define CC 1024
#define HH 4096
#define MM (PP*BB*TT)          // 16384 rows
#define LN_EPS 1e-5f

// ---------------------------------------------------------------- utilities

__global__ void f32_to_f16_kernel(const float* __restrict__ in,
                                  _Float16* __restrict__ out, int n) {
    int i = blockIdx.x * blockDim.x + threadIdx.x;
    if (i < n) out[i] = (_Float16)in[i];
}

__device__ __forceinline__ float block_sum(float v, float* sbuf) {
    int tid = threadIdx.x;
    sbuf[tid] = v;
    __syncthreads();
    #pragma unroll
    for (int s = 128; s > 0; s >>= 1) {
        if (tid < s) sbuf[tid] += sbuf[tid + s];
        __syncthreads();
    }
    float r = sbuf[0];
    __syncthreads();
    return r;
}

// --------------------------------------------- LayerNorm + time_shift + mix
// One block per row m = ((p*B+b)*T + t). Recomputes LN of row t-1 (xx).

__global__ void __launch_bounds__(256)
ln_mix_att_kernel(const float* __restrict__ x,
                  const float* __restrict__ g, const float* __restrict__ bta,
                  const float* __restrict__ mk, const float* __restrict__ mv,
                  const float* __restrict__ mr,
                  _Float16* __restrict__ xk, _Float16* __restrict__ xv,
                  _Float16* __restrict__ xr) {
    __shared__ float sbuf[256];
    const int m   = blockIdx.x;
    const int t   = m & (TT - 1);
    const int p   = m / (BB * TT);
    const int tid = threadIdx.x;
    const size_t row = (size_t)m * CC;

    float cur[4], prv[4];
    float s = 0.f, s2 = 0.f;
    #pragma unroll
    for (int i = 0; i < 4; ++i) {
        float v = x[row + tid + i * 256];
        cur[i] = v; s += v; s2 += v * v;
    }
    float mu  = block_sum(s,  sbuf) * (1.f / CC);
    float var = block_sum(s2, sbuf) * (1.f / CC) - mu * mu;
    float rs  = rsqrtf(var + LN_EPS);

    float mu_p = 0.f, rs_p = 0.f;
    if (t > 0) {
        s = 0.f; s2 = 0.f;
        #pragma unroll
        for (int i = 0; i < 4; ++i) {
            float v = x[row - CC + tid + i * 256];
            prv[i] = v; s += v; s2 += v * v;
        }
        mu_p = block_sum(s, sbuf) * (1.f / CC);
        float vp = block_sum(s2, sbuf) * (1.f / CC) - mu_p * mu_p;
        rs_p = rsqrtf(vp + LN_EPS);
    } else {
        #pragma unroll
        for (int i = 0; i < 4; ++i) prv[i] = 0.f;
    }

    #pragma unroll
    for (int i = 0; i < 4; ++i) {
        int c = tid + i * 256;
        float gg = g[c], bb = bta[c];
        float h  = (cur[i] - mu) * rs * gg + bb;
        float hh = (t > 0) ? ((prv[i] - mu_p) * rs_p * gg + bb) : 0.f;
        float k_ = mk[p * CC + c], v_ = mv[p * CC + c], r_ = mr[p * CC + c];
        xk[row + c] = (_Float16)(h * k_ + hh * (1.f - k_));
        xv[row + c] = (_Float16)(h * v_ + hh * (1.f - v_));
        xr[row + c] = (_Float16)(h * r_ + hh * (1.f - r_));
    }
}

__global__ void __launch_bounds__(256)
ln_mix_cm_kernel(const float* __restrict__ x,
                 const float* __restrict__ g, const float* __restrict__ bta,
                 const float* __restrict__ mk, const float* __restrict__ mr,
                 _Float16* __restrict__ xk, _Float16* __restrict__ xr) {
    __shared__ float sbuf[256];
    const int m   = blockIdx.x;
    const int t   = m & (TT - 1);
    const int p   = m / (BB * TT);
    const int tid = threadIdx.x;
    const size_t row = (size_t)m * CC;

    float cur[4], prv[4];
    float s = 0.f, s2 = 0.f;
    #pragma unroll
    for (int i = 0; i < 4; ++i) {
        float v = x[row + tid + i * 256];
        cur[i] = v; s += v; s2 += v * v;
    }
    float mu  = block_sum(s,  sbuf) * (1.f / CC);
    float var = block_sum(s2, sbuf) * (1.f / CC) - mu * mu;
    float rs  = rsqrtf(var + LN_EPS);

    float mu_p = 0.f, rs_p = 0.f;
    if (t > 0) {
        s = 0.f; s2 = 0.f;
        #pragma unroll
        for (int i = 0; i < 4; ++i) {
            float v = x[row - CC + tid + i * 256];
            prv[i] = v; s += v; s2 += v * v;
        }
        mu_p = block_sum(s, sbuf) * (1.f / CC);
        float vp = block_sum(s2, sbuf) * (1.f / CC) - mu_p * mu_p;
        rs_p = rsqrtf(vp + LN_EPS);
    } else {
        #pragma unroll
        for (int i = 0; i < 4; ++i) prv[i] = 0.f;
    }

    #pragma unroll
    for (int i = 0; i < 4; ++i) {
        int c = tid + i * 256;
        float gg = g[c], bb = bta[c];
        float h  = (cur[i] - mu) * rs * gg + bb;
        float hh = (t > 0) ? ((prv[i] - mu_p) * rs_p * gg + bb) : 0.f;
        float k_ = mk[p * CC + c], r_ = mr[p * CC + c];
        xk[row + c] = (_Float16)(h * k_ + hh * (1.f - k_));
        xr[row + c] = (_Float16)(h * r_ + hh * (1.f - r_));
    }
}

// ------------------------------------------------------------- WMMA GEMM
// D[m,n] = sum_k A[m,k] * W[n,k]   (A: MxK f16 row-major, W: NxK f16 row-major)
// Block = 256 threads = 8 waves laid out 4(M) x 2(N).
// Wave tile 32(M) x 64(N): 8 WMMAs / 12 b128 loads per k-step, with a manual
// double-buffered pipeline: loads of step k+1 are issued before the WMMAs of
// step k. __launch_bounds__(256,1) relaxes the occupancy target so all
// fragments (~160 VGPRs) stay live instead of being serialized.
//
// A fragment (16x32 f16, k order chosen here, not the ISA table's): lane l holds row (l&15);
//   lanes 0-15 -> K 0..15, lanes 16-31 -> K 16..31 (contiguous, matches B).
// B fragment (32x16 f16): lane l holds W-row (l&15), 16 contiguous K;
//   lanes 0-15 -> K 0..15, lanes 16-31 -> K 16..31.
// D (16x16 f32, 8 VGPRs): lane l col (l&15); VGPR i row = i + (l>=16 ? 8 : 0).

__device__ __forceinline__ v16h load_a_frag(const _Float16* __restrict__ p) {
    half8 lo = *(const half8*)(p);        // K chunk +0..7
    half8 hi = *(const half8*)(p + 8);    // K chunk +8..15
    v16h a;
    #pragma unroll
    for (int i = 0; i < 8; ++i) { a[i] = lo[i]; a[i + 8] = hi[i]; }
    return a;
}

#define WMMA_F16(ACC, AF, BF) \
    ACC = __builtin_amdgcn_wmma_f32_16x16x32_f16(false, AF, false, BF, \
                                                 (short)0, ACC, false, false)

// MODE: 0 = store f32, 1 = sigmoid f32, 2 = relu^2 -> f16, 3 = resid + acc f32
template<int MODE>
__global__ void __launch_bounds__(256, 1)
gemm_wmma_kernel(const _Float16* __restrict__ A, const _Float16* __restrict__ W,
                 float* __restrict__ outF, _Float16* __restrict__ outH,
                 const float* __restrict__ resid, int M, int N, int K) {
    const int lane = threadIdx.x & 31;
    const int wave = threadIdx.x >> 5;
    const int wm   = wave >> 1;           // 0..3
    const int wn   = wave & 1;            // 0..1
    const int m0   = blockIdx.y * 128 + wm * 32;
    const int n0   = blockIdx.x * 128 + wn * 64;
    const int l15  = lane & 15;
    const int lhi  = lane >> 4;           // 0 or 1

    const _Float16* a0p = A + (size_t)(m0 + l15) * K + lhi * 16;
    const _Float16* a1p = a0p + (size_t)16 * K;
    const _Float16* b0p = W + (size_t)(n0 + l15) * K + lhi * 16;
    const _Float16* b1p = b0p + (size_t)16 * K;
    const _Float16* b2p = b0p + (size_t)32 * K;
    const _Float16* b3p = b0p + (size_t)48 * K;

    v8f acc00 = {}, acc01 = {}, acc02 = {}, acc03 = {};
    v8f acc10 = {}, acc11 = {}, acc12 = {}, acc13 = {};

    // prologue: fragments for k0 = 0
    v16h a0 = load_a_frag(a0p);
    v16h a1 = load_a_frag(a1p);
    v16h b0 = *(const v16h*)(b0p);
    v16h b1 = *(const v16h*)(b1p);
    v16h b2 = *(const v16h*)(b2p);
    v16h b3 = *(const v16h*)(b3p);

    for (int k0 = 32; k0 < K; k0 += 32) {
        __builtin_prefetch(a0p + k0 + 224, 0, 3);  // global_prefetch_b8
        __builtin_prefetch(a1p + k0 + 224, 0, 3);
        __builtin_prefetch(b0p + k0 + 224, 0, 3);
        __builtin_prefetch(b2p + k0 + 224, 0, 3);
        // loads for step k0 issued before the WMMAs of step k0-32
        v16h na0 = load_a_frag(a0p + k0);
        v16h na1 = load_a_frag(a1p + k0);
        v16h nb0 = *(const v16h*)(b0p + k0);
        v16h nb1 = *(const v16h*)(b1p + k0);
        v16h nb2 = *(const v16h*)(b2p + k0);
        v16h nb3 = *(const v16h*)(b3p + k0);

        WMMA_F16(acc00, a0, b0);
        WMMA_F16(acc10, a1, b0);
        WMMA_F16(acc01, a0, b1);
        WMMA_F16(acc11, a1, b1);
        WMMA_F16(acc02, a0, b2);
        WMMA_F16(acc12, a1, b2);
        WMMA_F16(acc03, a0, b3);
        WMMA_F16(acc13, a1, b3);

        a0 = na0; a1 = na1;
        b0 = nb0; b1 = nb1; b2 = nb2; b3 = nb3;
    }
    // epilogue k-step
    WMMA_F16(acc00, a0, b0);
    WMMA_F16(acc10, a1, b0);
    WMMA_F16(acc01, a0, b1);
    WMMA_F16(acc11, a1, b1);
    WMMA_F16(acc02, a0, b2);
    WMMA_F16(acc12, a1, b2);
    WMMA_F16(acc03, a0, b3);
    WMMA_F16(acc13, a1, b3);

    v8f accs[2][4] = {{acc00, acc01, acc02, acc03},
                      {acc10, acc11, acc12, acc13}};
    #pragma unroll
    for (int mt = 0; mt < 2; ++mt) {
        #pragma unroll
        for (int nt = 0; nt < 4; ++nt) {
            const int col = n0 + nt * 16 + l15;
            #pragma unroll
            for (int i = 0; i < 8; ++i) {
                const int rrow = m0 + mt * 16 + lhi * 8 + i;
                const size_t idx = (size_t)rrow * N + col;
                const float v = accs[mt][nt][i];
                if (MODE == 0) {
                    outF[idx] = v;
                } else if (MODE == 1) {
                    outF[idx] = 1.f / (1.f + __expf(-v));
                } else if (MODE == 2) {
                    float r = v > 0.f ? v : 0.f;
                    outH[idx] = (_Float16)(r * r);
                } else {
                    outF[idx] = resid[idx] + v;
                }
            }
        }
    }
}

// ------------------------------------------------------------------ WKV scan
// One thread per (batch-row, channel). Writes sy = sigmoid(r) * y as f16,
// ready to feed the Wo GEMM.

__global__ void __launch_bounds__(256)
wkv_kernel(const float* __restrict__ k, const float* __restrict__ v,
           const float* __restrict__ sr,
           const float* __restrict__ time_decay,
           const float* __restrict__ time_first,
           _Float16* __restrict__ sy) {
    const int idx = blockIdx.x * blockDim.x + threadIdx.x;
    if (idx >= PP * BB * CC) return;
    const int c  = idx & (CC - 1);
    const int nb = idx / CC;

    const float w = -__expf(time_decay[c]);
    const float u = time_first[c];
    float aa = 0.f, bb = 0.f, pp = -1e38f;
    const size_t base = (size_t)nb * TT * CC + c;

    for (int t = 0; t < TT; ++t) {
        const size_t off = base + (size_t)t * CC;
        const float kk = k[off];
        const float vv = v[off];
        float ww = u + kk;
        float p  = fmaxf(pp, ww);
        float e1 = __expf(pp - p);
        float e2 = __expf(ww - p);
        float y  = (e1 * aa + e2 * vv) / (e1 * bb + e2);
        float ww2 = w + pp;
        float p2  = fmaxf(ww2, kk);
        float e1b = __expf(ww2 - p2);
        float e2b = __expf(kk - p2);
        aa = e1b * aa + e2b * vv;
        bb = e1b * bb + e2b;
        pp = p2;
        sy[off] = (_Float16)(sr[off] * y);
    }
}

// ------------------------------------------------------------ final combine

__global__ void __launch_bounds__(256)
combine_kernel(const float* __restrict__ x1, const float* __restrict__ sr2,
               const float* __restrict__ kv, float* __restrict__ out, size_t n) {
    size_t i = (size_t)blockIdx.x * blockDim.x + threadIdx.x;
    if (i < n) out[i] = x1[i] + sr2[i] * kv[i];
}

// ---------------------------------------------------------------- launcher

extern "C" void kernel_launch(void* const* d_in, const int* in_sizes, int n_in,
                              void* d_out, int out_size, void* d_ws, size_t ws_size,
                              hipStream_t stream) {
    (void)in_sizes; (void)n_in; (void)out_size; (void)ws_size;

    const float* x    = (const float*)d_in[0];
    const float* ln1g = (const float*)d_in[1];
    const float* ln1b = (const float*)d_in[2];
    const float* ln2g = (const float*)d_in[3];
    const float* ln2b = (const float*)d_in[4];
    const float* td   = (const float*)d_in[5];
    const float* tf   = (const float*)d_in[6];
    const float* amk  = (const float*)d_in[7];
    const float* amv  = (const float*)d_in[8];
    const float* amr  = (const float*)d_in[9];
    const float* Wk   = (const float*)d_in[10];
    const float* Wv   = (const float*)d_in[11];
    const float* Wr   = (const float*)d_in[12];
    const float* Wo   = (const float*)d_in[13];
    const float* cmk  = (const float*)d_in[14];
    const float* cmr  = (const float*)d_in[15];
    const float* Wck  = (const float*)d_in[16];
    const float* Wcr  = (const float*)d_in[17];
    const float* Wcv  = (const float*)d_in[18];
    float* out = (float*)d_out;

    // ---- workspace layout (256B aligned; peak ~380 MiB with aliasing) ----
    char* ws = (char*)d_ws;
    size_t off = 0;
    auto take = [&](size_t bytes) -> char* {
        char* r = ws + off;
        off += (bytes + 255) & ~(size_t)255;
        return r;
    };
    const size_t MC = (size_t)MM * CC;   // 16,777,216
    const size_t WC = (size_t)CC * CC;   // 1,048,576
    const size_t WH = (size_t)HH * CC;   // 4,194,304

    _Float16* wk16  = (_Float16*)take(WC * 2);
    _Float16* wv16  = (_Float16*)take(WC * 2);
    _Float16* wr16  = (_Float16*)take(WC * 2);
    _Float16* wo16  = (_Float16*)take(WC * 2);
    _Float16* wck16 = (_Float16*)take(WH * 2);
    _Float16* wcr16 = (_Float16*)take(WC * 2);
    _Float16* wcv16 = (_Float16*)take(WH * 2);
    _Float16* bufA0 = (_Float16*)take(MC * 2);  // xk  -> sy  -> sr2(lo half)
    _Float16* bufA1 = (_Float16*)take(MC * 2);  // xv  -> xk2 -> sr2(hi half)
    _Float16* bufA2 = (_Float16*)take(MC * 2);  // xr  -> xr2
    float*    bufF0 = (float*)take(MC * 4);     // k   -> x1
    float*    bufF1 = (float*)take(MC * 4);     // v   -> kk(lo half, f16)
    float*    bufF2 = (float*)take(MC * 4);     // sr  -> kk(hi half, f16)
    float*    bufG  = (float*)take(MC * 4);     // kv
    float*    sr2   = (float*)bufA0;            // 64 MiB overlay of A0+A1
    _Float16* kkBuf = (_Float16*)bufF1;         // 128 MiB overlay of F1+F2

    // ---- 1: weights -> f16 ----
    f32_to_f16_kernel<<<WC / 256, 256, 0, stream>>>(Wk,  wk16,  (int)WC);
    f32_to_f16_kernel<<<WC / 256, 256, 0, stream>>>(Wv,  wv16,  (int)WC);
    f32_to_f16_kernel<<<WC / 256, 256, 0, stream>>>(Wr,  wr16,  (int)WC);
    f32_to_f16_kernel<<<WC / 256, 256, 0, stream>>>(Wo,  wo16,  (int)WC);
    f32_to_f16_kernel<<<WH / 256, 256, 0, stream>>>(Wck, wck16, (int)WH);
    f32_to_f16_kernel<<<WC / 256, 256, 0, stream>>>(Wcr, wcr16, (int)WC);
    f32_to_f16_kernel<<<WH / 256, 256, 0, stream>>>(Wcv, wcv16, (int)WH);

    // ---- 2: LN1 + shift + mix -> xk, xv, xr (f16) ----
    ln_mix_att_kernel<<<MM, 256, 0, stream>>>(x, ln1g, ln1b, amk, amv, amr,
                                              bufA0, bufA1, bufA2);

    // ---- 3: k / v / sigmoid(r) GEMMs ----
    dim3 gC(CC / 128, MM / 128);
    gemm_wmma_kernel<0><<<gC, 256, 0, stream>>>(bufA0, wk16, bufF0, nullptr,
                                                nullptr, MM, CC, CC);
    gemm_wmma_kernel<0><<<gC, 256, 0, stream>>>(bufA1, wv16, bufF1, nullptr,
                                                nullptr, MM, CC, CC);
    gemm_wmma_kernel<1><<<gC, 256, 0, stream>>>(bufA2, wr16, bufF2, nullptr,
                                                nullptr, MM, CC, CC);

    // ---- 4: WKV scan; fuses sigmoid(r)*y, emits f16 ----
    wkv_kernel<<<(PP * BB * CC) / 256, 256, 0, stream>>>(bufF0, bufF1, bufF2,
                                                         td, tf, bufA0);

    // ---- 5: x1 = x + sy @ Wo^T ----
    gemm_wmma_kernel<3><<<gC, 256, 0, stream>>>(bufA0, wo16, bufF0, nullptr,
                                                x, MM, CC, CC);

    // ---- 6: LN2 + shift + mix -> xk2, xr2 (f16) ----
    ln_mix_cm_kernel<<<MM, 256, 0, stream>>>(bufF0, ln2g, ln2b, cmk, cmr,
                                             bufA1, bufA2);

    // ---- 7: kk = relu(xk2 @ Wck^T)^2  (f16, M x H) ----
    dim3 gH(HH / 128, MM / 128);
    gemm_wmma_kernel<2><<<gH, 256, 0, stream>>>(bufA1, wck16, nullptr, kkBuf,
                                                nullptr, MM, HH, CC);

    // ---- 8: sr2 = sigmoid(xr2 @ Wcr^T) ----
    gemm_wmma_kernel<1><<<gC, 256, 0, stream>>>(bufA2, wcr16, sr2, nullptr,
                                                nullptr, MM, CC, CC);

    // ---- 9: kv = kk @ Wcv^T  (K = H) ----
    gemm_wmma_kernel<0><<<gC, 256, 0, stream>>>(kkBuf, wcv16, bufG, nullptr,
                                                nullptr, MM, CC, HH);

    // ---- 10: out = x1 + sr2 * kv ----
    combine_kernel<<<(unsigned)(MC / 256), 256, 0, stream>>>(bufF0, sr2, bufG,
                                                             out, MC);
}
